// HTAPBiasAttention_26826365731320
// MI455X (gfx1250) — matrix operand order repaired and checked by reading, not compiled, not run on hardware
//
#include <hip/hip_runtime.h>
#include <hip/hip_bf16.h>

// ---------------------------------------------------------------------------
// HTAP bias attention for MI455X (gfx1250, wave32, WMMA)
// B=16 N=256 HID=512 H=8 DK=64 FEAT=8 MLP_H=64 LAM=0.1 SCALE=1/8
// All matmul stages on v_wmma_f32_16x16x32_f16.
// ---------------------------------------------------------------------------

typedef float     v8f  __attribute__((ext_vector_type(8)));
typedef _Float16  v8h  __attribute__((ext_vector_type(8)));
typedef _Float16  v16h __attribute__((ext_vector_type(16)));

__device__ __forceinline__ v16h cat8(v8h a, v8h b) {
  return __builtin_shufflevector(a, b, 0,1,2,3,4,5,6,7,8,9,10,11,12,13,14,15);
}

// A fragment (16x32 f16, M=lane%16): halves K = kk+hi*8..+7 and kk+16+hi*8..+7
__device__ __forceinline__ v16h ldA16(const _Float16* A, int lda, int row0, int kk, int lane) {
  int lo = lane & 15, hi = lane >> 4;
  const _Float16* p = A + (size_t)(row0 + lo) * lda + kk + hi * 8;
  return cat8(*(const v8h*)p, *(const v8h*)(p + 16));
}
// B fragment (32x16 f16, N=lane%16) from K-contiguous (transposed) storage:
// halves K = kk+hi*8..+7 and kk+16+hi*8..+7
__device__ __forceinline__ v16h ldB16(const _Float16* Bt, int ldb, int col0, int kk, int lane) {
  int lo = lane & 15, hi = lane >> 4;
  const _Float16* p = Bt + (size_t)(col0 + lo) * ldb + kk + hi * 8;
  return cat8(*(const v8h*)p, *(const v8h*)(p + 16));
}
__device__ __forceinline__ v8f wmma16(v16h a, v16h b, v8f c) {
  return __builtin_amdgcn_wmma_f32_16x16x32_f16(false, a, false, b, (short)0, c, false, false);
}

// ---------------------------------------------------------------------------
// 1) fp32 -> f16 cast, and transposed-cast of weight matrices
// ---------------------------------------------------------------------------
__global__ void cast_f16_k(const float* __restrict__ src, _Float16* __restrict__ dst, int n) {
  int i = blockIdx.x * blockDim.x + threadIdx.x;
  if (i < n) dst[i] = (_Float16)src[i];
}
__global__ void transpose_f16_k(const float* __restrict__ W, _Float16* __restrict__ WT) {
  int i = blockIdx.x * blockDim.x + threadIdx.x;  // 512*512
  int n = i >> 9, k = i & 511;
  WT[(size_t)n * 512 + k] = (_Float16)W[(size_t)k * 512 + n];
}

// ---------------------------------------------------------------------------
// 2) per-node MLP terms: hA[s][bn][u] = feat_s[bn]@W1_s[:8], hB uses W1_s[8:16]
// ---------------------------------------------------------------------------
__global__ void node_mlp_k(const float* __restrict__ fs, const float* __restrict__ fo,
                           const float* __restrict__ w1s, const float* __restrict__ w1o,
                           float* __restrict__ hA, float* __restrict__ hB) {
  int bn = blockIdx.x;   // 0..4095
  int u  = threadIdx.x;  // 0..63
  const float* feats[2] = { fs, fo };
  const float* w1[2]    = { w1s, w1o };
  for (int s = 0; s < 2; ++s) {
    float a = 0.f, b = 0.f;
    #pragma unroll
    for (int f = 0; f < 8; ++f) {
      float fv = feats[s][(size_t)bn * 8 + f];
      a += fv * w1[s][f * 64 + u];
      b += fv * w1[s][(8 + f) * 64 + u];
    }
    hA[(size_t)s * (4096 * 64) + (size_t)bn * 64 + u] = a;
    hB[(size_t)s * (4096 * 64) + (size_t)bn * 64 + u] = b;
  }
}

// ---------------------------------------------------------------------------
// 3) projection GEMM: out = (X@W + bias) * scale, X f16 [4096,512], WT f16 [512,512]
//    mode 0: out f16 [B,H,N,DK]   (q,k heads)
//    mode 1: out f16 [B,H,DK,N]   (v transposed for attn@v B-fragments)
// ---------------------------------------------------------------------------
__global__ void proj_gemm_f16out_k(const _Float16* __restrict__ X, const _Float16* __restrict__ WT,
                                   const float* __restrict__ bias, float scale,
                                   _Float16* __restrict__ out, int mode) {
  int bid = blockIdx.x;                  // 64 row-blocks x 8 col-blocks
  int rb = bid >> 3, cb = bid & 7;
  int tid = threadIdx.x, lane = tid & 31, wave = tid >> 5;
  int lo = lane & 15, hi = lane >> 4;
  int row0 = rb * 64 + (wave >> 1) * 16;
  int col0 = cb * 64 + (wave & 1) * 32;
  v8f c0 = {}, c1 = {};
  for (int kk = 0; kk < 512; kk += 32) {
    v16h a  = ldA16(X, 512, row0, kk, lane);
    v16h b0 = ldB16(WT, 512, col0, kk, lane);
    v16h b1 = ldB16(WT, 512, col0 + 16, kk, lane);
    c0 = wmma16(a, b0, c0);
    c1 = wmma16(a, b1, c1);
  }
  for (int t = 0; t < 2; ++t) {
    v8f c = t ? c1 : c0;
    int col = col0 + t * 16 + lo;
    float bv = bias[col];
    int hh = col >> 6, dk = col & 63;
    #pragma unroll
    for (int v = 0; v < 8; ++v) {
      int g = row0 + v + 8 * hi;
      int bz = g >> 8, n = g & 255;
      float val = (c[v] + bv) * scale;
      size_t idx = (mode == 0)
        ? ((size_t)((bz * 8 + hh) * 256 + n)) * 64 + dk
        : ((size_t)((bz * 8 + hh) * 64 + dk)) * 256 + n;
      out[idx] = (_Float16)val;
    }
  }
}

// output projection: x f16 [4096,512] @ WoT + bo -> f32 [4096,512]
__global__ void proj_gemm_f32out_k(const _Float16* __restrict__ X, const _Float16* __restrict__ WT,
                                   const float* __restrict__ bias, float* __restrict__ out) {
  int bid = blockIdx.x;
  int rb = bid >> 3, cb = bid & 7;
  int tid = threadIdx.x, lane = tid & 31, wave = tid >> 5;
  int lo = lane & 15, hi = lane >> 4;
  int row0 = rb * 64 + (wave >> 1) * 16;
  int col0 = cb * 64 + (wave & 1) * 32;
  v8f c0 = {}, c1 = {};
  for (int kk = 0; kk < 512; kk += 32) {
    v16h a  = ldA16(X, 512, row0, kk, lane);
    v16h b0 = ldB16(WT, 512, col0, kk, lane);
    v16h b1 = ldB16(WT, 512, col0 + 16, kk, lane);
    c0 = wmma16(a, b0, c0);
    c1 = wmma16(a, b1, c1);
  }
  for (int t = 0; t < 2; ++t) {
    v8f c = t ? c1 : c0;
    int col = col0 + t * 16 + lo;
    float bv = bias[col];
    #pragma unroll
    for (int v = 0; v < 8; ++v) {
      int g = row0 + v + 8 * hi;
      out[(size_t)g * 512 + col] = c[v] + bv;
    }
  }
}

// ---------------------------------------------------------------------------
// 4) pairwise MLP bias via f16 WMMA (16x16x32, K zero-padded): htap[b,i,j,h]
//    block = (b, i-tile16, j-tile16); 4 waves, each wave 4 i-rows.
//    layer1: D = |f_i-f_j|@Wc + C(=hA_i+hB_j+b1), relu -> LDS f16
//    layer2: D = h@W2 + C(=b2s+b2o), 2 K-steps of 32
// ---------------------------------------------------------------------------
__global__ void pair_bias_k(const float* __restrict__ fs, const float* __restrict__ fo,
                            const float* __restrict__ w1s, const float* __restrict__ w1o,
                            const float* __restrict__ b1s, const float* __restrict__ b1o,
                            const float* __restrict__ w2s, const float* __restrict__ w2o,
                            const float* __restrict__ b2s, const float* __restrict__ b2o,
                            const float* __restrict__ hA, const float* __restrict__ hB,
                            float* __restrict__ htap) {
  __shared__ float s_fi[2][16][8], s_fj[2][16][8];
  __shared__ float s_hA[2][16][64], s_hB[2][16][64];
  __shared__ __align__(16) _Float16 s_WcT[2][64][32];  // [s][u][f], f>=8 zero
  __shared__ __align__(16) _Float16 s_W2T[2][8][64];   // [s][t][u]
  __shared__ float s_b1[2][64];
  __shared__ float s_b2[8];
  __shared__ __align__(16) _Float16 s_h16[4][2][16][64];  // per-wave hidden (relu'd)

  int jt = blockIdx.x, it = blockIdx.y, bz = blockIdx.z;
  int i0 = it * 16, j0 = jt * 16;
  int tid = threadIdx.x, lane = tid & 31, wave = tid >> 5;
  int lo = lane & 15, hi = lane >> 4;

  const float* feats[2] = { fs, fo };
  const float* w1[2]    = { w1s, w1o };
  const float* w2[2]    = { w2s, w2o };
  const float* b1[2]    = { b1s, b1o };

  for (int idx = tid; idx < 256; idx += 128) {
    int s = idx >> 7, rem = idx & 127, r = rem >> 3, f = rem & 7;
    s_fi[s][r][f] = feats[s][(size_t)(bz * 256 + i0 + r) * 8 + f];
    s_fj[s][r][f] = feats[s][(size_t)(bz * 256 + j0 + r) * 8 + f];
  }
  for (int idx = tid; idx < 2048; idx += 128) {
    int s = idx >> 10, rem = idx & 1023, r = rem >> 6, u = rem & 63;
    s_hA[s][r][u] = hA[(size_t)s * (4096 * 64) + (size_t)(bz * 256 + i0 + r) * 64 + u];
    s_hB[s][r][u] = hB[(size_t)s * (4096 * 64) + (size_t)(bz * 256 + j0 + r) * 64 + u];
  }
  for (int idx = tid; idx < 4096; idx += 128) {  // WcT zero-padded
    int s = idx >> 11, rem = idx & 2047, u = rem >> 5, f = rem & 31;
    s_WcT[s][u][f] = (f < 8) ? (_Float16)w1[s][(16 + f) * 64 + u] : (_Float16)0.f;
  }
  for (int idx = tid; idx < 1024; idx += 128) {
    int s = idx >> 9, rem = idx & 511, t = rem >> 6, u = rem & 63;
    s_W2T[s][t][u] = (_Float16)w2[s][u * 8 + t];
  }
  for (int idx = tid; idx < 128; idx += 128) {
    int s = idx >> 6, u = idx & 63;
    s_b1[s][u] = b1[s][u];
  }
  if (tid < 8) s_b2[tid] = b2s[tid] + b2o[tid];
  __syncthreads();

  for (int rr = 0; rr < 4; ++rr) {
    int r = wave * 4 + rr;  // i-row within tile
    for (int s = 0; s < 2; ++s) {
      // A fragment: rows M = pair (j-offset), K = 8 features zero-padded to 32.
      // Slot map: hi==0 slots0-7 <-> K0-7 (real); all other slots are K>=8 -> 0.
      v16h a = {};
      if (hi == 0) {
        #pragma unroll
        for (int f = 0; f < 8; ++f)
          a[f] = (_Float16)fabsf(s_fi[s][r][f] - s_fj[s][lo][f]);
      }
      #pragma unroll
      for (int u0 = 0; u0 < 64; u0 += 16) {
        v8f c;
        #pragma unroll
        for (int v = 0; v < 8; ++v)
          c[v] = s_hA[s][r][u0 + lo] + s_hB[s][v + 8 * hi][u0 + lo] + s_b1[s][u0 + lo];
        // B fragment from s_WcT: col u = u0+lo, halves K = hi*16..+15
        {
          const _Float16* p = &s_WcT[s][u0 + lo][hi * 16];
          v16h bfr = cat8(*(const v8h*)p, *(const v8h*)(p + 8));
          c = wmma16(a, bfr, c);
        }
        #pragma unroll
        for (int v = 0; v < 8; ++v)
          s_h16[wave][s][v + 8 * hi][u0 + lo] = (_Float16)fmaxf(c[v], 0.f);
      }
    }
    // second layer: d = h_s@W2s + h_o@W2o + (b2s+b2o); cols 8..15 dead
    v8f d;
    #pragma unroll
    for (int v = 0; v < 8; ++v) d[v] = (lo < 8) ? s_b2[lo] : 0.f;
    int t7 = lo & 7;
    #pragma unroll
    for (int s = 0; s < 2; ++s) {
      #pragma unroll
      for (int kk = 0; kk < 64; kk += 32) {
        const _Float16* pa = &s_h16[wave][s][lo][kk + hi * 8];
        v16h a2 = cat8(*(const v8h*)pa, *(const v8h*)(pa + 16));
        const _Float16* pb = &s_W2T[s][t7][kk + hi * 8];
        v16h b2 = cat8(*(const v8h*)pb, *(const v8h*)(pb + 16));
        d = wmma16(a2, b2, d);
      }
    }
    if (lo < 8) {
      #pragma unroll
      for (int v = 0; v < 8; ++v) {
        int j = j0 + v + 8 * hi;
        htap[((size_t)(bz * 256 + i0 + r) * 256 + j) * 8 + lo] = d[v];
      }
    }
  }
}

// ---------------------------------------------------------------------------
// 5) attention: scores WMMA + biases -> softmax (LDS) -> attn@V WMMA -> x f16
//    block = (n-tile16, h, b); 4 waves
// ---------------------------------------------------------------------------
__global__ void attn_k(const _Float16* __restrict__ qh, const _Float16* __restrict__ kh,
                       const _Float16* __restrict__ vhT, const float* __restrict__ tb,
                       const float* __restrict__ htap, _Float16* __restrict__ xo) {
  __shared__ float s_sc[16][256];
  __shared__ __align__(16) _Float16 s_at[16][256];
  __shared__ float s_red[16][16];

  int nt = blockIdx.x, hh = blockIdx.y, bz = blockIdx.z;
  int n0 = nt * 16;
  int tid = threadIdx.x, lane = tid & 31, wave = tid >> 5;
  int lo = lane & 15, hi = lane >> 4;

  const _Float16* qb = qh + (size_t)((bz * 8 + hh) * 256) * 64;
  const _Float16* kb = kh + (size_t)((bz * 8 + hh) * 256) * 64;

  // scores: 16 x 256 ; q A-fragments are m-invariant -> hoist
  v16h a0 = ldA16(qb, 64, n0, 0, lane);
  v16h a1 = ldA16(qb, 64, n0, 32, lane);
  for (int q = 0; q < 4; ++q) {
    int mt = wave * 4 + q;
    v8f c = {};
    c = wmma16(a0, ldB16(kb, 64, mt * 16, 0, lane), c);
    c = wmma16(a1, ldB16(kb, 64, mt * 16, 32, lane), c);
    int m = mt * 16 + lo;
    #pragma unroll
    for (int v = 0; v < 8; ++v) {
      int n = n0 + v + 8 * hi;
      float sc = c[v]
        + tb[((size_t)((bz * 8 + hh) * 256 + n)) * 256 + m]
        + 0.1f * htap[((size_t)(bz * 256 + n) * 256 + m) * 8 + hh];
      s_sc[v + 8 * hi][m] = sc;
    }
  }
  __syncthreads();

  // softmax over 256 cols, 8 threads per row
  int row = tid >> 3, c8 = tid & 7;
  float mx = -1e30f;
  for (int k = c8; k < 256; k += 8) mx = fmaxf(mx, s_sc[row][k]);
  s_red[row][c8] = mx;
  __syncthreads();
  if (c8 == 0) {
    float m2 = s_red[row][0];
    for (int t = 1; t < 8; ++t) m2 = fmaxf(m2, s_red[row][t]);
    s_red[row][8] = m2;
  }
  __syncthreads();
  float rm = s_red[row][8];
  float sum = 0.f;
  for (int k = c8; k < 256; k += 8) {
    float e = __expf(s_sc[row][k] - rm);
    s_sc[row][k] = e;
    sum += e;
  }
  s_red[row][c8] = sum;
  __syncthreads();
  if (c8 == 0) {
    float s2 = 0.f;
    for (int t = 0; t < 8; ++t) s2 += s_red[row][t];
    s_red[row][9] = s2;
  }
  __syncthreads();
  float inv = 1.f / s_red[row][9];
  for (int k = c8; k < 256; k += 8) s_at[row][k] = (_Float16)(s_sc[row][k] * inv);
  __syncthreads();

  // attn (16x256) @ vh (256x64): wave w -> d-tile w
  {
    int dt = wave;
    v8f c = {};
    const _Float16* vb = vhT + (size_t)((bz * 8 + hh) * 64) * 256;
    for (int kk = 0; kk < 256; kk += 32) {
      v8h alo = *(const v8h*)&s_at[lo][kk + hi * 8];
      v8h ahi = *(const v8h*)&s_at[lo][kk + 16 + hi * 8];
      v16h a  = cat8(alo, ahi);
      v16h bb = ldB16(vb, 256, dt * 16, kk, lane);
      c = wmma16(a, bb, c);
    }
    int col = hh * 64 + dt * 16 + lo;
    #pragma unroll
    for (int v = 0; v < 8; ++v) {
      int n = n0 + v + 8 * hi;
      xo[(size_t)(bz * 256 + n) * 512 + col] = (_Float16)c[v];
    }
  }
}

// ---------------------------------------------------------------------------
extern "C" void kernel_launch(void* const* d_in, const int* in_sizes, int n_in,
                              void* d_out, int out_size, void* d_ws, size_t ws_size,
                              hipStream_t stream) {
  (void)in_sizes; (void)n_in; (void)out_size; (void)ws_size;
  const float* q    = (const float*)d_in[0];
  const float* k    = (const float*)d_in[1];
  const float* v    = (const float*)d_in[2];
  const float* tb   = (const float*)d_in[3];
  const float* fs   = (const float*)d_in[4];
  const float* fo   = (const float*)d_in[5];
  const float* Wq   = (const float*)d_in[6];  const float* bq = (const float*)d_in[7];
  const float* Wk   = (const float*)d_in[8];  const float* bk = (const float*)d_in[9];
  const float* Wv   = (const float*)d_in[10]; const float* bv = (const float*)d_in[11];
  const float* Wo   = (const float*)d_in[12]; const float* bo = (const float*)d_in[13];
  const float* fsW1 = (const float*)d_in[14]; const float* fsb1 = (const float*)d_in[15];
  const float* fsW2 = (const float*)d_in[16]; const float* fsb2 = (const float*)d_in[17];
  const float* foW1 = (const float*)d_in[18]; const float* fob1 = (const float*)d_in[19];
  const float* foW2 = (const float*)d_in[20]; const float* fob2 = (const float*)d_in[21];

  char* ws = (char*)d_ws;
  size_t off = 0;
  auto alloc = [&](size_t bytes) -> char* {
    char* p = ws + off;
    off += (bytes + 255) & ~(size_t)255;
    return p;
  };
  const size_t NE = 4096 * 512;  // rows x HID
  _Float16* q16  = (_Float16*)alloc(NE * 2);
  _Float16* k16  = (_Float16*)alloc(NE * 2);
  _Float16* v16  = (_Float16*)alloc(NE * 2);
  _Float16* WqT  = (_Float16*)alloc(512 * 512 * 2);
  _Float16* WkT  = (_Float16*)alloc(512 * 512 * 2);
  _Float16* WvT  = (_Float16*)alloc(512 * 512 * 2);
  _Float16* WoT  = (_Float16*)alloc(512 * 512 * 2);
  _Float16* qhp  = (_Float16*)alloc(NE * 2);   // [B,H,N,DK]
  _Float16* khp  = (_Float16*)alloc(NE * 2);   // [B,H,N,DK]
  _Float16* vhT  = (_Float16*)alloc(NE * 2);   // [B,H,DK,N]
  _Float16* x16  = (_Float16*)alloc(NE * 2);   // [B,N,HID]
  float*    hA   = (float*)alloc(2ull * 4096 * 64 * 4);
  float*    hB   = (float*)alloc(2ull * 4096 * 64 * 4);
  float*    htap = (float*)alloc(16ull * 256 * 256 * 8 * 4);

  cast_f16_k<<<8192, 256, 0, stream>>>(q, q16, (int)NE);
  cast_f16_k<<<8192, 256, 0, stream>>>(k, k16, (int)NE);
  cast_f16_k<<<8192, 256, 0, stream>>>(v, v16, (int)NE);
  transpose_f16_k<<<1024, 256, 0, stream>>>(Wq, WqT);
  transpose_f16_k<<<1024, 256, 0, stream>>>(Wk, WkT);
  transpose_f16_k<<<1024, 256, 0, stream>>>(Wv, WvT);
  transpose_f16_k<<<1024, 256, 0, stream>>>(Wo, WoT);
  node_mlp_k<<<4096, 64, 0, stream>>>(fs, fo, fsW1, foW1, hA, hB);

  proj_gemm_f16out_k<<<512, 256, 0, stream>>>(q16, WqT, bq, 0.125f, qhp, 0);
  proj_gemm_f16out_k<<<512, 256, 0, stream>>>(k16, WkT, bk, 1.0f,   khp, 0);
  proj_gemm_f16out_k<<<512, 256, 0, stream>>>(v16, WvT, bv, 1.0f,   vhT, 1);

  pair_bias_k<<<dim3(16, 16, 16), 128, 0, stream>>>(fs, fo, fsW1, foW1, fsb1, fob1,
                                                    fsW2, foW2, fsb2, fob2, hA, hB, htap);

  attn_k<<<dim3(16, 8, 16), 128, 0, stream>>>(qhp, khp, vhT, tb, htap, x16);

  proj_gemm_f32out_k<<<512, 256, 0, stream>>>(x16, WoT, bo, (float*)d_out);
}
